// GraphEncoder_85676007620916
// MI455X (gfx1250) — matrix operand order repaired and checked by reading, not compiled, not run on hardware
//
#include <hip/hip_runtime.h>
#include <cstddef>
#include <cstdint>

typedef __attribute__((ext_vector_type(16))) _Float16 v16h;
typedef __attribute__((ext_vector_type(8)))  _Float16 v8h;
typedef __attribute__((ext_vector_type(4)))  _Float16 v4h;
typedef __attribute__((ext_vector_type(8)))  float    v8f;
typedef __attribute__((ext_vector_type(4)))  float    v4f;
typedef __attribute__((ext_vector_type(4)))  int      v4i;

#define Ncells 6144
#define NIN    2000
#define NINP   2016    // K-pad of 2000 to multiple of 32
#define NINP2  2048    // N-pad of 2000 to multiple of 64
#define E1d    256
#define E2d    64
#define D1d    64
#define D2d    256
#define NZd    32
#define NZP    64
#define NCd    10
#define NCP    64

enum { EPI_NONE = 0, EPI_BIAS, EPI_BIAS_RELU, EPI_BIAS_RELU_BN,
       EPI_BIAS_SIGMOID, EPI_BIAS_EXPCLIP, EPI_BIAS_SOFTPLUS, EPI_ELU };

// ---------------------------------------------------------------------------
// padded f32 -> f16 convert: dst[r*ld+c] = (r<srcR && c<srcC) ? src : 0
// ---------------------------------------------------------------------------
__global__ __launch_bounds__(256) void pad_conv_kernel(const float* __restrict__ src,
                                                       _Float16* __restrict__ dst,
                                                       int srcR, int srcC, int dstR, int ld) {
  int idx = blockIdx.x * 256 + threadIdx.x;
  if (idx < dstR * ld) {
    int r = idx / ld, c = idx % ld;
    float v = (r < srcR && c < srcC) ? src[(size_t)r * srcC + c] : 0.0f;
    dst[idx] = (_Float16)v;
  }
}

// strided residual mix: o[r*ldo+c] = 0.6*a[r*lda+c] + 0.4*b[r*ldb+c]
__global__ __launch_bounds__(256) void mix_kernel(const float* __restrict__ a, int lda,
                                                  const float* __restrict__ b, int ldb,
                                                  _Float16* __restrict__ o, int ldo,
                                                  int rows, int cols) {
  int idx = blockIdx.x * 256 + threadIdx.x;
  if (idx < rows * cols) {
    int r = idx / cols, c = idx % cols;
    o[(size_t)r * ldo + c] =
        (_Float16)(0.6f * a[(size_t)r * lda + c] + 0.4f * b[(size_t)r * ldb + c]);
  }
}

// ---------------------------------------------------------------------------
// WMMA GEMM: C[6144,NdimPad] = A[6144,K]@B[K,NdimPad], f16 in / f32 accum.
// Tile-aligned dims (M%64==0, NdimPad%64==0, K%32==0); staging uses CDNA5
// async global->LDS (GLOBAL_LOAD_ASYNC_TO_LDS_B128, ASYNCcnt) with double-
// buffered LDS: issue next tile async, compute WMMA on current, then
// s_wait_asynccnt + barrier. 128 threads = 4 waves, 64x64 tile.
// ---------------------------------------------------------------------------
template <int EPI>
__global__ __launch_bounds__(128) void wmma_gemm(
    const _Float16* __restrict__ A, const _Float16* __restrict__ B,
    float* __restrict__ Cf, _Float16* __restrict__ Ch,
    const float* __restrict__ bias, const float* __restrict__ bng,
    const float* __restrict__ bnb, const float* __restrict__ bnm,
    const float* __restrict__ bnv,
    int Nstore, int Kdim, int lda, int ldb, int ldc) {
  __shared__ _Float16 As[2][64][40];   // 64x32 tiles, row stride 80B
  __shared__ _Float16 Bs[2][32][72];   // 32x64 tiles, row stride 144B

  const int tid  = threadIdx.x;
  const int lane = tid & 31, wid = tid >> 5;
  const int wr = wid >> 1, wc = wid & 1;
  const int lm = lane & 15, half = lane >> 4;
  const size_t tm = (size_t)blockIdx.y * 64, tn = (size_t)blockIdx.x * 64;

  // staging coordinates: 2 x 16B chunks per thread per operand
  const int ar0 = tid >> 2, ac0 = (tid & 3) * 8;   // A rows 0..31 (+32)
  const int br0 = tid >> 3, bc0 = (tid & 7) * 8;   // B rows 0..15 (+16)

  // wave-relative LDS byte offsets of this thread's staging chunks
  unsigned la0[2], la1[2], lb0[2], lb1[2];
  #pragma unroll
  for (int b = 0; b < 2; ++b) {
    la0[b] = (unsigned)(uintptr_t)&As[b][ar0][ac0];
    la1[b] = (unsigned)(uintptr_t)&As[b][ar0 + 32][ac0];
    lb0[b] = (unsigned)(uintptr_t)&Bs[b][br0][bc0];
    lb1[b] = (unsigned)(uintptr_t)&Bs[b][br0 + 16][bc0];
  }

  auto stage = [&](int k0, int buf) {
    unsigned long long Ag =
        (unsigned long long)(uintptr_t)(A + (tm + ar0) * lda + (k0 + ac0));
    unsigned long long Bg =
        (unsigned long long)(uintptr_t)(B + (size_t)(k0 + br0) * ldb + tn + bc0);
    unsigned long long Ag2 = Ag + (unsigned long long)32 * lda * 2;
    unsigned long long Bg2 = Bg + (unsigned long long)16 * ldb * 2;
    asm volatile("global_load_async_to_lds_b128 %0, %1, off"
                 :: "v"(la0[buf]), "v"(Ag) : "memory");
    asm volatile("global_load_async_to_lds_b128 %0, %1, off"
                 :: "v"(la1[buf]), "v"(Ag2) : "memory");
    asm volatile("global_load_async_to_lds_b128 %0, %1, off"
                 :: "v"(lb0[buf]), "v"(Bg) : "memory");
    asm volatile("global_load_async_to_lds_b128 %0, %1, off"
                 :: "v"(lb1[buf]), "v"(Bg2) : "memory");
  };

  v8f acc[2][2] = {};

  stage(0, 0);
  asm volatile("s_wait_asynccnt 0x0" ::: "memory");
  __syncthreads();

  int buf = 0;
  for (int k0 = 0; k0 < Kdim; k0 += 32) {
    const bool more = (k0 + 32) < Kdim;
    if (more) stage(k0 + 32, buf ^ 1);     // prefetch next tile into other buffer

    v16h afr[2], bfr[2];
    // A frag: lane lm carries row M0; its 16 K-values are two contiguous runs
    #pragma unroll
    for (int mi = 0; mi < 2; ++mi) {
      int M0 = wr * 32 + mi * 16 + lm;
      v8h lo = *(const v8h*)&As[buf][M0][half * 8];
      v8h hi = *(const v8h*)&As[buf][M0][16 + half * 8];
      #pragma unroll
      for (int e = 0; e < 8; ++e) { afr[mi][e] = lo[e]; afr[mi][8 + e] = hi[e]; }
    }
    // B frag: column gather (N = lm), K pairs across rows
    #pragma unroll
    for (int ni = 0; ni < 2; ++ni) {
      int Nc = wc * 32 + ni * 16 + lm;
      #pragma unroll
      for (int i = 0; i < 8; ++i) {
        int k = half * 8 + 2 * i + (i >= 4 ? 8 : 0);
        bfr[ni][2 * i]     = Bs[buf][k][Nc];
        bfr[ni][2 * i + 1] = Bs[buf][k + 1][Nc];
      }
    }
    #pragma unroll
    for (int mi = 0; mi < 2; ++mi)
      #pragma unroll
      for (int ni = 0; ni < 2; ++ni)
        acc[mi][ni] = __builtin_amdgcn_wmma_f32_16x16x32_f16(
            false, afr[mi], false, bfr[ni], (short)0, acc[mi][ni], false, false);

    if (more) {
      asm volatile("s_wait_asynccnt 0x0" ::: "memory");
      __syncthreads();
      buf ^= 1;
    }
  }

  // epilogue: VGPR r holds M = r + 8*half, N = lm; per-column params hoisted
  #pragma unroll
  for (int ni = 0; ni < 2; ++ni) {
    int col = (int)tn + wc * 32 + ni * 16 + lm;
    if (col < Nstore) {
      float bcol = 0.0f, scl = 1.0f, sft = 0.0f;
      if constexpr (EPI == EPI_BIAS || EPI == EPI_BIAS_RELU || EPI == EPI_BIAS_RELU_BN ||
                    EPI == EPI_BIAS_SIGMOID || EPI == EPI_BIAS_EXPCLIP ||
                    EPI == EPI_BIAS_SOFTPLUS)
        bcol = bias[col];
      if constexpr (EPI == EPI_BIAS_RELU_BN) {
        scl = rsqrtf(bnv[col] + 1e-5f) * bng[col];
        sft = bnb[col] - bnm[col] * scl;
      }
      #pragma unroll
      for (int mi = 0; mi < 2; ++mi) {
        #pragma unroll
        for (int r = 0; r < 8; ++r) {
          size_t row = tm + wr * 32 + mi * 16 + half * 8 + r;
          float v = acc[mi][ni][r];
          if constexpr (EPI == EPI_BIAS) {
            v += bcol;
          } else if constexpr (EPI == EPI_BIAS_RELU) {
            v = fmaxf(v + bcol, 0.0f);
          } else if constexpr (EPI == EPI_BIAS_RELU_BN) {
            v = fmaxf(v + bcol, 0.0f) * scl + sft;
          } else if constexpr (EPI == EPI_BIAS_SIGMOID) {
            v = 1.0f / (1.0f + __expf(-(v + bcol)));
          } else if constexpr (EPI == EPI_BIAS_EXPCLIP) {
            v = fminf(fmaxf(__expf(v + bcol), 1e-5f), 1e4f);
          } else if constexpr (EPI == EPI_BIAS_SOFTPLUS) {
            float t = v + bcol;
            float sp = (t > 20.0f) ? t : log1pf(__expf(t));
            v = fminf(fmaxf(sp, 1e-4f), 1e3f);
          } else if constexpr (EPI == EPI_ELU) {
            v = (v > 0.0f) ? v : (__expf(v) - 1.0f);
          }
          Cf[row * ldc + col] = v;
          if (Ch) Ch[row * ldc + col] = (_Float16)v;
        }
      }
    }
  }
}

// ---------------------------------------------------------------------------
// as = h @ a_s, an = h @ a_n (one thread per row; h has leading dim ld)
// ---------------------------------------------------------------------------
__global__ __launch_bounds__(256) void gat_scores(const float* __restrict__ h, int ld,
                                                  const float* __restrict__ a_s,
                                                  const float* __restrict__ a_n,
                                                  float* __restrict__ asv,
                                                  float* __restrict__ anv,
                                                  int n, int d) {
  int i = blockIdx.x * 256 + threadIdx.x;
  if (i < n) {
    float s = 0.0f, t = 0.0f;
    for (int k = 0; k < d; ++k) {
      float hv = h[(size_t)i * ld + k];
      s += hv * a_s[k];
      t += hv * a_n[k];
    }
    asv[i] = s;
    anv[i] = t;
  }
}

// ---------------------------------------------------------------------------
// One block per row: logits -> leaky -> mask -> softmax -> f16 att row.
// Row (24 KB) lives in LDS; N x N M/adj traffic is b128-vectorized.
// ---------------------------------------------------------------------------
__global__ __launch_bounds__(256) void gat_attention(const float* __restrict__ asv,
                                                     const float* __restrict__ anv,
                                                     const float* __restrict__ Mw,
                                                     const int* __restrict__ adj,
                                                     _Float16* __restrict__ att) {
  __shared__ float row[Ncells];
  __shared__ float red[256];
  const int i = blockIdx.x;
  const int tid = threadIdx.x;
  const float ai = asv[i];
  const size_t base = (size_t)i * Ncells;

  float lmax = -3.0e38f;
  for (int j = tid * 4; j < Ncells; j += 1024) {
    v4f m  = *(const v4f*)(Mw + base + j);
    v4i ad = *(const v4i*)(adj + base + j);
    v4f an = *(const v4f*)(anv + j);
    #pragma unroll
    for (int e = 0; e < 4; ++e) {
      float ev = (ai + an[e]) * m[e];
      ev = (ev > 0.0f) ? ev : 0.2f * ev;
      if (ad[e] <= 0) ev = -9.0e15f;
      row[j + e] = ev;
      lmax = fmaxf(lmax, ev);
    }
  }
  red[tid] = lmax;
  __syncthreads();
  for (int s = 128; s > 0; s >>= 1) {
    if (tid < s) red[tid] = fmaxf(red[tid], red[tid + s]);
    __syncthreads();
  }
  const float gmax = red[0];
  __syncthreads();

  float lsum = 0.0f;
  for (int j = tid * 4; j < Ncells; j += 1024) {
    #pragma unroll
    for (int e = 0; e < 4; ++e) {
      float ex = __expf(row[j + e] - gmax);
      row[j + e] = ex;
      lsum += ex;
    }
  }
  red[tid] = lsum;
  __syncthreads();
  for (int s = 128; s > 0; s >>= 1) {
    if (tid < s) red[tid] += red[tid + s];
    __syncthreads();
  }
  const float inv = 1.0f / red[0];
  __syncthreads();

  for (int j = tid * 4; j < Ncells; j += 1024) {
    v4h o;
    #pragma unroll
    for (int e = 0; e < 4; ++e) o[e] = (_Float16)(row[j + e] * inv);
    *(v4h*)(att + base + j) = o;
  }
}

// ---------------------------------------------------------------------------
// t = h1^T @ h4  (256x10) ; h4 has leading dim h4ld
// ---------------------------------------------------------------------------
__global__ __launch_bounds__(256) void t_kernel(const float* __restrict__ h1,
                                                const float* __restrict__ h4, int h4ld,
                                                float* __restrict__ t) {
  int idx = blockIdx.x * 256 + threadIdx.x;
  if (idx < E1d * NCd) {
    int c = idx / NCd, j = idx % NCd;
    float s = 0.0f;
    for (int i = 0; i < Ncells; ++i)
      s += h1[(size_t)i * E1d + c] * h4[(size_t)i * h4ld + j];
    t[idx] = s;
  }
}

// z1 = h1 @ t  (re-associated s@h4 = h1@(h1^T@h4))
__global__ __launch_bounds__(256) void z1_kernel(const float* __restrict__ h1,
                                                 const float* __restrict__ t,
                                                 float* __restrict__ z1) {
  int idx = blockIdx.x * 256 + threadIdx.x;
  if (idx < Ncells * NCd) {
    int i = idx / NCd, j = idx % NCd;
    float s = 0.0f;
    for (int c = 0; c < E1d; ++c)
      s += h1[(size_t)i * E1d + c] * t[c * NCd + j];
    z1[idx] = s;
  }
}

__global__ __launch_bounds__(256) void predict_kernel(const float* __restrict__ z1,
                                                      float* __restrict__ out) {
  int i = blockIdx.x * 256 + threadIdx.x;
  if (i < Ncells) {
    float s = 0.0f;
    for (int j = 0; j < NCd; ++j) {
      float v = z1[i * NCd + j];
      s += v * v;
    }
    float inv = 1.0f / fmaxf(sqrtf(s), 1e-12f);
    for (int j = 0; j < NCd; ++j) out[i * NCd + j] = z1[i * NCd + j] * inv;
  }
}

// DEC soft assignment q + deterministic per-block column partial sums
__global__ __launch_bounds__(256) void q_kernel(const float* __restrict__ z,
                                                const float* __restrict__ cluster,
                                                float* __restrict__ q_out,
                                                float* __restrict__ partial) {
  __shared__ float sq[256 * NCd];
  int i = blockIdx.x * 256 + threadIdx.x;
  float qr[NCd];
  if (i < Ncells) {
    float rs = 0.0f;
    for (int j = 0; j < NCd; ++j) {
      float d = 0.0f;
      for (int k = 0; k < NZd; ++k) {
        float t = z[(size_t)i * NZd + k] - cluster[j * NZd + k];
        d += t * t;
      }
      float qv = 1.0f / (1.0f + d);
      qv = qv * qv;                 // /2 cancels in row normalization
      qr[j] = qv;
      rs += qv;
    }
    float inv = 1.0f / rs;
    for (int j = 0; j < NCd; ++j) {
      qr[j] *= inv;
      q_out[(size_t)i * NCd + j] = qr[j];
    }
  } else {
    for (int j = 0; j < NCd; ++j) qr[j] = 0.0f;
  }
  for (int j = 0; j < NCd; ++j) sq[threadIdx.x * NCd + j] = qr[j];
  __syncthreads();
  if (threadIdx.x < NCd) {
    float s = 0.0f;
    for (int k = 0; k < 256; ++k) s += sq[k * NCd + threadIdx.x];
    partial[blockIdx.x * NCd + threadIdx.x] = s;
  }
}

__global__ __launch_bounds__(32) void colsum_kernel(const float* __restrict__ partial,
                                                    float* __restrict__ colsum, int nb) {
  int j = threadIdx.x;
  if (j < NCd) {
    float s = 0.0f;
    for (int b = 0; b < nb; ++b) s += partial[b * NCd + j];
    colsum[j] = s;
  }
}

__global__ __launch_bounds__(256) void p_kernel(const float* __restrict__ q,
                                                const float* __restrict__ colsum,
                                                float* __restrict__ p) {
  int i = blockIdx.x * 256 + threadIdx.x;
  if (i < Ncells) {
    float w[NCd], rs = 0.0f;
    for (int j = 0; j < NCd; ++j) {
      float t = q[(size_t)i * NCd + j];
      w[j] = t * t / colsum[j];
      rs += w[j];
    }
    float inv = 1.0f / rs;
    for (int j = 0; j < NCd; ++j) p[(size_t)i * NCd + j] = w[j] * inv;
  }
}

// ---------------------------------------------------------------------------
// host orchestration
// ---------------------------------------------------------------------------
static inline dim3 gemm_grid(int m, int npad) { return dim3(npad / 64, m / 64); }

extern "C" void kernel_launch(void* const* d_in, const int* in_sizes, int n_in,
                              void* d_out, int out_size, void* d_ws, size_t ws_size,
                              hipStream_t stream) {
  (void)in_sizes; (void)n_in; (void)out_size; (void)ws_size;
  const int N = Ncells;

  // ---- inputs ----
  const float* x      = (const float*)d_in[0];
  const int*   adj    = (const int*)  d_in[1];
  const float* Mw     = (const float*)d_in[2];
  const float* W_enc1 = (const float*)d_in[3];  const float* b_enc1 = (const float*)d_in[4];
  const float* bn1_g  = (const float*)d_in[5];  const float* bn1_b  = (const float*)d_in[6];
  const float* bn1_m  = (const float*)d_in[7];  const float* bn1_v  = (const float*)d_in[8];
  const float* W_enc2 = (const float*)d_in[9];  const float* b_enc2 = (const float*)d_in[10];
  const float* bn2_g  = (const float*)d_in[11]; const float* bn2_b  = (const float*)d_in[12];
  const float* bn2_m  = (const float*)d_in[13]; const float* bn2_v  = (const float*)d_in[14];
  const float* W_z    = (const float*)d_in[15]; const float* b_z    = (const float*)d_in[16];
  const float* W_dec1 = (const float*)d_in[17]; const float* b_dec1 = (const float*)d_in[18];
  const float* W_dec2 = (const float*)d_in[19]; const float* b_dec2 = (const float*)d_in[20];
  const float* W_pi   = (const float*)d_in[21]; const float* b_pi   = (const float*)d_in[22];
  const float* W_mean = (const float*)d_in[23]; const float* b_mean = (const float*)d_in[24];
  const float* W_disp = (const float*)d_in[25]; const float* b_disp = (const float*)d_in[26];
  const float* g1W = (const float*)d_in[27]; const float* g1s = (const float*)d_in[28]; const float* g1n = (const float*)d_in[29];
  const float* g2W = (const float*)d_in[30]; const float* g2s = (const float*)d_in[31]; const float* g2n = (const float*)d_in[32];
  const float* g3W = (const float*)d_in[33]; const float* g3s = (const float*)d_in[34]; const float* g3n = (const float*)d_in[35];
  const float* g4W = (const float*)d_in[36]; const float* g4s = (const float*)d_in[37]; const float* g4n = (const float*)d_in[38];
  const float* cluster = (const float*)d_in[39];

  // ---- outputs (flat, return order) ----
  float* out      = (float*)d_out;
  float* z_out    = out;
  float* pi_out   = z_out    + (size_t)N * NZd;
  float* mean_out = pi_out   + (size_t)N * NIN;
  float* disp_out = mean_out + (size_t)N * NIN;
  float* pred_out = disp_out + (size_t)N * NIN;
  float* q_out    = pred_out + (size_t)N * NCd;
  float* p_out    = q_out    + (size_t)N * NCd;

  // ---- workspace allocator ----
  char* wsb = (char*)d_ws;
  size_t off = 0;
  auto alloc = [&](size_t bytes) -> void* {
    void* p = wsb + off;
    off += (bytes + 255) & ~(size_t)255;
    return p;
  };
  _Float16* xh       = (_Float16*)alloc((size_t)N * NINP * 2);
  _Float16* wench1h  = (_Float16*)alloc((size_t)NINP * E1d * 2);
  _Float16* wench2h  = (_Float16*)alloc((size_t)E1d * E2d * 2);
  _Float16* wzh      = (_Float16*)alloc((size_t)E2d * NZP * 2);
  _Float16* wdec1h   = (_Float16*)alloc((size_t)NZd * D1d * 2);
  _Float16* wdec2h   = (_Float16*)alloc((size_t)D1d * D2d * 2);
  _Float16* wpih     = (_Float16*)alloc((size_t)D2d * NINP2 * 2);
  _Float16* wmeanh   = (_Float16*)alloc((size_t)D2d * NINP2 * 2);
  _Float16* wdisph   = (_Float16*)alloc((size_t)D2d * NINP2 * 2);
  _Float16* g1Wh     = (_Float16*)alloc((size_t)NINP * E1d * 2);
  _Float16* g2Wh     = (_Float16*)alloc((size_t)E1d * E2d * 2);
  _Float16* g3Wh     = (_Float16*)alloc((size_t)E2d * NZP * 2);
  _Float16* g4Wh     = (_Float16*)alloc((size_t)NZd * NCP * 2);
  float*    ench1_f  = (float*)   alloc((size_t)N * E1d * 4);
  _Float16* ench1_h  = (_Float16*)alloc((size_t)N * E1d * 2);
  float*    ench2_f  = (float*)   alloc((size_t)N * E2d * 4);
  _Float16* ench2_h  = (_Float16*)alloc((size_t)N * E2d * 2);
  _Float16* zh       = (_Float16*)alloc((size_t)N * NZd * 2);
  _Float16* dech1_h  = (_Float16*)alloc((size_t)N * D1d * 2);
  _Float16* dech2_h  = (_Float16*)alloc((size_t)N * D2d * 2);
  float*    scratchf = (float*)   alloc((size_t)N * D2d * 4);
  float*    hpre_f   = (float*)   alloc((size_t)N * E1d * 4);
  _Float16* hpre_h   = (_Float16*)alloc((size_t)N * E1d * 2);
  float*    asv      = (float*)   alloc((size_t)N * 4);
  float*    anv      = (float*)   alloc((size_t)N * 4);
  _Float16* att      = (_Float16*)alloc((size_t)N * N * 2);    // 75.5 MB
  float*    h1_f     = (float*)   alloc((size_t)N * E1d * 4);
  float*    h2_f     = (float*)   alloc((size_t)N * E2d * 4);
  float*    h3_f     = (float*)   alloc((size_t)N * NZP * 4);
  float*    h4_f     = (float*)   alloc((size_t)N * NCP * 4);
  _Float16* mix_h    = (_Float16*)alloc((size_t)N * E1d * 2);
  float*    t_buf    = (float*)   alloc((size_t)E1d * NCd * 4);
  float*    z1_buf   = (float*)   alloc((size_t)N * NCd * 4);
  float*    partial  = (float*)   alloc((size_t)24 * NCd * 4);
  float*    colsum   = (float*)   alloc((size_t)NCd * 4);

  auto pad = [&](const float* s, _Float16* d, int sr, int sc, int dr, int ld) {
    int tot = dr * ld;
    pad_conv_kernel<<<dim3((tot + 255) / 256), 256, 0, stream>>>(s, d, sr, sc, dr, ld);
  };

  // ---- zero-padded f16 operand prep ----
  pad(x,      xh,      N,    NIN,  N,    NINP);
  pad(W_enc1, wench1h, NIN,  E1d,  NINP, E1d);
  pad(W_enc2, wench2h, E1d,  E2d,  E1d,  E2d);
  pad(W_z,    wzh,     E2d,  NZd,  E2d,  NZP);
  pad(W_dec1, wdec1h,  NZd,  D1d,  NZd,  D1d);
  pad(W_dec2, wdec2h,  D1d,  D2d,  D1d,  D2d);
  pad(W_pi,   wpih,    D2d,  NIN,  D2d,  NINP2);
  pad(W_mean, wmeanh,  D2d,  NIN,  D2d,  NINP2);
  pad(W_disp, wdisph,  D2d,  NIN,  D2d,  NINP2);
  pad(g1W,    g1Wh,    NIN,  E1d,  NINP, E1d);
  pad(g2W,    g2Wh,    E1d,  E2d,  E1d,  E2d);
  pad(g3W,    g3Wh,    E2d,  NZd,  E2d,  NZP);
  pad(g4W,    g4Wh,    NZd,  NCd,  NZd,  NCP);

  // ---- autoencoder ----
  wmma_gemm<EPI_BIAS_RELU_BN><<<gemm_grid(N, E1d), 128, 0, stream>>>(
      xh, wench1h, ench1_f, ench1_h, b_enc1, bn1_g, bn1_b, bn1_m, bn1_v,
      E1d, NINP, NINP, E1d, E1d);
  wmma_gemm<EPI_BIAS_RELU_BN><<<gemm_grid(N, E2d), 128, 0, stream>>>(
      ench1_h, wench2h, ench2_f, ench2_h, b_enc2, bn2_g, bn2_b, bn2_m, bn2_v,
      E2d, E1d, E1d, E2d, E2d);
  wmma_gemm<EPI_BIAS><<<gemm_grid(N, NZP), 128, 0, stream>>>(
      ench2_h, wzh, z_out, zh, b_z, nullptr, nullptr, nullptr, nullptr,
      NZd, E2d, E2d, NZP, NZd);
  wmma_gemm<EPI_BIAS_RELU><<<gemm_grid(N, D1d), 128, 0, stream>>>(
      zh, wdec1h, scratchf, dech1_h, b_dec1, nullptr, nullptr, nullptr, nullptr,
      D1d, NZd, NZd, D1d, D1d);
  wmma_gemm<EPI_BIAS_RELU><<<gemm_grid(N, D2d), 128, 0, stream>>>(
      dech1_h, wdec2h, scratchf, dech2_h, b_dec2, nullptr, nullptr, nullptr, nullptr,
      D2d, D1d, D1d, D2d, D2d);
  wmma_gemm<EPI_BIAS_SIGMOID><<<gemm_grid(N, NINP2), 128, 0, stream>>>(
      dech2_h, wpih, pi_out, nullptr, b_pi, nullptr, nullptr, nullptr, nullptr,
      NIN, D2d, D2d, NINP2, NIN);
  wmma_gemm<EPI_BIAS_EXPCLIP><<<gemm_grid(N, NINP2), 128, 0, stream>>>(
      dech2_h, wmeanh, mean_out, nullptr, b_mean, nullptr, nullptr, nullptr, nullptr,
      NIN, D2d, D2d, NINP2, NIN);
  wmma_gemm<EPI_BIAS_SOFTPLUS><<<gemm_grid(N, NINP2), 128, 0, stream>>>(
      dech2_h, wdisph, disp_out, nullptr, b_disp, nullptr, nullptr, nullptr, nullptr,
      NIN, D2d, D2d, NINP2, NIN);

  // ---- GAT stack ----
  auto gat = [&](const _Float16* Ain, int K, int lda, const _Float16* Wh, int dPad,
                 int dAct, const float* a_s, const float* a_n, float* hout_f) {
    wmma_gemm<EPI_NONE><<<gemm_grid(N, dPad), 128, 0, stream>>>(
        Ain, Wh, hpre_f, hpre_h, nullptr, nullptr, nullptr, nullptr, nullptr,
        dPad, K, lda, dPad, dPad);
    gat_scores<<<dim3(N / 256), 256, 0, stream>>>(hpre_f, dPad, a_s, a_n, asv, anv, N, dAct);
    gat_attention<<<dim3(N), 256, 0, stream>>>(asv, anv, Mw, adj, att);
    wmma_gemm<EPI_ELU><<<gemm_grid(N, dPad), 128, 0, stream>>>(
        att, hpre_h, hout_f, nullptr, nullptr, nullptr, nullptr, nullptr, nullptr,
        dPad, N, N, dPad, dPad);
  };

  gat(xh, NINP, NINP, g1Wh, E1d, E1d, g1s, g1n, h1_f);
  mix_kernel<<<dim3((N * E1d + 255) / 256), 256, 0, stream>>>(
      h1_f, E1d, ench1_f, E1d, mix_h, E1d, N, E1d);
  gat(mix_h, E1d, E1d, g2Wh, E2d, E2d, g2s, g2n, h2_f);
  mix_kernel<<<dim3((N * E2d + 255) / 256), 256, 0, stream>>>(
      h2_f, E2d, ench2_f, E2d, mix_h, E2d, N, E2d);
  gat(mix_h, E2d, E2d, g3Wh, NZP, NZd, g3s, g3n, h3_f);
  mix_kernel<<<dim3((N * NZd + 255) / 256), 256, 0, stream>>>(
      h3_f, NZP, z_out, NZd, mix_h, NZd, N, NZd);
  gat(mix_h, NZd, NZd, g4Wh, NCP, NCd, g4s, g4n, h4_f);

  // ---- predict: z1 = (h1 h1^T) h4 = h1 (h1^T h4) ----
  t_kernel<<<dim3((E1d * NCd + 255) / 256), 256, 0, stream>>>(h1_f, h4_f, NCP, t_buf);
  z1_kernel<<<dim3((N * NCd + 255) / 256), 256, 0, stream>>>(h1_f, t_buf, z1_buf);
  predict_kernel<<<dim3(N / 256), 256, 0, stream>>>(z1_buf, pred_out);

  // ---- DEC q / p ----
  q_kernel<<<dim3(N / 256), 256, 0, stream>>>(z_out, cluster, q_out, partial);
  colsum_kernel<<<dim3(1), 32, 0, stream>>>(partial, colsum, N / 256);
  p_kernel<<<dim3(N / 256), 256, 0, stream>>>(q_out, colsum, p_out);
}
